// OpticalLinear_68934225101140
// MI455X (gfx1250) — matrix operand order repaired and checked by reading, not compiled, not run on hardware
//
#include <hip/hip_runtime.h>
#include <hip/hip_bf16.h>

// ---------------------------------------------------------------------------
// OpticalLinear (lumina_nano_v1) for MI455X / gfx1250
//   out = quant4( (quant4(x) @ W^T) * 0.85 * (1-0.05) )
// A-side is exact: quant4(x) = n * scale_x with n in [-15,15]; we carry n in
// f16 (exact) through v_wmma_f32_16x16x32_f16 and fold scale_x*0.8075 into
// the epilogue. Per-tensor abs-max reductions use bit-pattern atomicMax.
//
// GEMM: block tile 256(M) x 128(N), 8 wave32's arranged 4(M) x 2(N); each
// wave owns a 64x64 region = 4x4 tiles of 16x16 -> 16 b128 loads per
// 16 v_wmma per K-step (1.0 loads/wmma), 16 independent WMMAs in flight.
// ---------------------------------------------------------------------------

typedef __attribute__((ext_vector_type(16))) _Float16 v16h;
typedef __attribute__((ext_vector_type(8)))  _Float16 v8h;
typedef __attribute__((ext_vector_type(4)))  _Float16 v4h;
typedef __attribute__((ext_vector_type(8)))  float    v8f;

static constexpr long MTOK = 8192;   // tokens (GEMM M)
static constexpr long KF   = 4096;   // in_features (GEMM K)
static constexpr long NF   = 4096;   // out_features (GEMM N)
static constexpr float ATT = 0.85f * (1.0f - 0.05f);   // 0.8075

union Frag16 { v16h v; v8h h[2]; };

// --------------------------------------------------------------------------
// 0) zero the two stat slots (ws is poisoned 0xAA by the harness)
// --------------------------------------------------------------------------
__global__ void init_stats_kernel(unsigned int* __restrict__ s) {
  if (threadIdx.x == 0) { s[0] = 0u; s[1] = 0u; }
}

// --------------------------------------------------------------------------
// 1) per-tensor abs-max of x (float4 grid-stride, LDS tree, bitwise atomicMax)
// --------------------------------------------------------------------------
__global__ void absmax_kernel(const float4* __restrict__ x, long n4,
                              unsigned int* __restrict__ out) {
  float m = 0.0f;
  long i      = (long)blockIdx.x * blockDim.x + threadIdx.x;
  long stride = (long)gridDim.x * blockDim.x;
  for (; i < n4; i += stride) {
    float4 v = x[i];
    m = fmaxf(m, fmaxf(fmaxf(fabsf(v.x), fabsf(v.y)),
                       fmaxf(fabsf(v.z), fabsf(v.w))));
  }
  __shared__ float red[256];
  red[threadIdx.x] = m;
  __syncthreads();
  for (int s = 128; s > 0; s >>= 1) {
    if (threadIdx.x < s)
      red[threadIdx.x] = fmaxf(red[threadIdx.x], red[threadIdx.x + s]);
    __syncthreads();
  }
  if (threadIdx.x == 0) atomicMax(out, __float_as_uint(red[0]));
}

// --------------------------------------------------------------------------
// 2a) quantize x to integer grid, store integers exactly in f16
// --------------------------------------------------------------------------
__global__ void quantize_x_kernel(const float4* __restrict__ x,
                                  v4h* __restrict__ a, long n4,
                                  const unsigned int* __restrict__ stats) {
  float mx    = __uint_as_float(stats[0]);
  float scale = mx * (1.0f / 15.0f);
  float inv   = (scale > 0.0f) ? (1.0f / scale) : 1.0f;
  long i      = (long)blockIdx.x * blockDim.x + threadIdx.x;
  long stride = (long)gridDim.x * blockDim.x;
  for (; i < n4; i += stride) {
    float4 v = x[i];
    v4h q;
    q.x = (_Float16)rintf(v.x * inv);   // |q| <= 15, exact in f16
    q.y = (_Float16)rintf(v.y * inv);
    q.z = (_Float16)rintf(v.z * inv);
    q.w = (_Float16)rintf(v.w * inv);
    a[i] = q;
  }
}

// --------------------------------------------------------------------------
// 2b) convert W f32 -> f16
// --------------------------------------------------------------------------
__global__ void convert_w_kernel(const float4* __restrict__ w,
                                 v4h* __restrict__ b, long n4) {
  long i      = (long)blockIdx.x * blockDim.x + threadIdx.x;
  long stride = (long)gridDim.x * blockDim.x;
  for (; i < n4; i += stride) {
    float4 v = w[i];
    v4h h;
    h.x = (_Float16)v.x;
    h.y = (_Float16)v.y;
    h.z = (_Float16)v.z;
    h.w = (_Float16)v.w;
    b[i] = h;
  }
}

// --------------------------------------------------------------------------
// 3) WMMA GEMM: C[m,n] = sum_k A[m,k] * B[n,k]   (A,B f16; C f32 raw integers)
//    Fragment addressing follows CDNA5 ISA VGPR layouts:
//      A 16x32: lane L: m = L%16; hi = L/16; K chunks [hi*8, hi*8+8) and +16
//      B 32x16: lane L: n = L%16; K chunks [hi*8, hi*8+8) and +16 (W is [N,K] row-major)
//    Also reduces global max|C| for the output quantizer.
// --------------------------------------------------------------------------
__global__ void __launch_bounds__(256)
gemm_wmma_kernel(const _Float16* __restrict__ A, const _Float16* __restrict__ B,
                 float* __restrict__ C, unsigned int* __restrict__ maxbits) {
  const int lane = threadIdx.x & 31;
  const int wid  = threadIdx.x >> 5;
  const int lr   = lane & 15;
  const int hi   = lane >> 4;
  const int wm   = wid & 3;    // 0..3 -> M offset in block (x64)
  const int wn   = wid >> 2;   // 0..1 -> N offset in block (x64)

  const long rowM0 = (long)blockIdx.y * 256 + (long)wm * 64;
  const long rowN0 = (long)blockIdx.x * 128 + (long)wn * 64;

  const _Float16* aP[4];
  const _Float16* bP[4];
#pragma unroll
  for (int i = 0; i < 4; ++i)
    aP[i] = A + (rowM0 + i * 16 + lr) * KF + hi * 8;
#pragma unroll
  for (int j = 0; j < 4; ++j)
    bP[j] = B + (rowN0 + j * 16 + lr) * KF + hi * 8;

  v8f acc[4][4];
#pragma unroll
  for (int i = 0; i < 4; ++i)
#pragma unroll
    for (int j = 0; j < 4; ++j)
      acc[i][j] = v8f{};

  for (int k = 0; k < (int)KF; k += 32) {
    Frag16 fa[4];
    Frag16 fb[4];
#pragma unroll
    for (int i = 0; i < 4; ++i) {
      fa[i].h[0] = *(const v8h*)(aP[i] + k);        // K = hi*8 .. +8
      fa[i].h[1] = *(const v8h*)(aP[i] + k + 16);   // K = hi*8+16 .. +8
    }
#pragma unroll
    for (int j = 0; j < 4; ++j) {
      fb[j].h[0] = *(const v8h*)(bP[j] + k);        // K = hi*8 .. +8
      fb[j].h[1] = *(const v8h*)(bP[j] + k + 16);   // K = hi*8+16 .. +8
    }
#pragma unroll
    for (int i = 0; i < 4; ++i)
#pragma unroll
      for (int j = 0; j < 4; ++j)
        acc[i][j] = __builtin_amdgcn_wmma_f32_16x16x32_f16(
            /*neg_a=*/false, fa[i].v, /*neg_b=*/false, fb[j].v,
            /*c_mod=*/(short)0, acc[i][j], /*reuse_a=*/false, /*reuse_b=*/false);
  }

  // epilogue: store raw integer-valued sums + track abs-max
  float m = 0.0f;
#pragma unroll
  for (int i = 0; i < 4; ++i)
#pragma unroll
    for (int j = 0; j < 4; ++j) {
      const long col = rowN0 + j * 16 + lr;
#pragma unroll
      for (int r = 0; r < 8; ++r) {
        const long row = rowM0 + i * 16 + hi * 8 + r;  // C/D layout: M=r(+8 hi)
        const float v = acc[i][j][r];
        C[row * NF + col] = v;
        m = fmaxf(m, fabsf(v));
      }
    }

  __shared__ float red[256];
  red[threadIdx.x] = m;
  __syncthreads();
  for (int s = 128; s > 0; s >>= 1) {
    if (threadIdx.x < s)
      red[threadIdx.x] = fmaxf(red[threadIdx.x], red[threadIdx.x + s]);
    __syncthreads();
  }
  if (threadIdx.x == 0) atomicMax(maxbits, __float_as_uint(red[0]));
}

// --------------------------------------------------------------------------
// 4) epilogue: noisy = C * scale_x * 0.8075 ; out = quant4(noisy), in place.
//    quant4(noisy) = rint(C * 15/maxC) * (maxC * s / 15)
// --------------------------------------------------------------------------
__global__ void finalize_kernel(float4* __restrict__ c, long n4,
                                const unsigned int* __restrict__ stats) {
  float mx_x    = __uint_as_float(stats[0]);
  float scale_x = mx_x * (1.0f / 15.0f);
  if (!(scale_x > 0.0f)) scale_x = 1.0f;
  const float s    = scale_x * ATT;          // fold attenuation + drift
  const float maxC = __uint_as_float(stats[1]);
  float inv, so;
  if (maxC > 0.0f) { inv = 15.0f / maxC; so = maxC * s * (1.0f / 15.0f); }
  else             { inv = 0.0f;         so = 1.0f; }

  long i      = (long)blockIdx.x * blockDim.x + threadIdx.x;
  long stride = (long)gridDim.x * blockDim.x;
  for (; i < n4; i += stride) {
    float4 v = c[i];
    v.x = rintf(v.x * inv) * so;
    v.y = rintf(v.y * inv) * so;
    v.z = rintf(v.z * inv) * so;
    v.w = rintf(v.w * inv) * so;
    c[i] = v;
  }
}

// --------------------------------------------------------------------------
extern "C" void kernel_launch(void* const* d_in, const int* in_sizes, int n_in,
                              void* d_out, int out_size, void* d_ws, size_t ws_size,
                              hipStream_t stream) {
  const float* x = (const float*)d_in[0];   // [8192, 4096] f32
  const float* w = (const float*)d_in[1];   // [4096, 4096] f32
  float* out     = (float*)d_out;           // [8192, 4096] f32

  // ws layout: [0..1] uint stats (absmax_x bits, absmax_C bits),
  //            +256B: A16 (64 MB), then W16 (32 MB)
  unsigned int* stats = (unsigned int*)d_ws;
  _Float16* A16 = (_Float16*)((char*)d_ws + 256);
  _Float16* W16 = A16 + MTOK * KF;

  const long nx4 = MTOK * KF / 4;   // 8,388,608 float4
  const long nw4 = NF * KF / 4;     // 4,194,304 float4
  const long nc4 = MTOK * NF / 4;   // 8,388,608 float4

  init_stats_kernel<<<1, 32, 0, stream>>>(stats);

  absmax_kernel<<<2048, 256, 0, stream>>>((const float4*)x, nx4, stats + 0);

  quantize_x_kernel<<<8192, 256, 0, stream>>>((const float4*)x, (v4h*)A16, nx4,
                                              stats);
  convert_w_kernel<<<8192, 256, 0, stream>>>((const float4*)w, (v4h*)W16, nw4);

  dim3 grid(NF / 128, MTOK / 256);  // (32, 32)
  gemm_wmma_kernel<<<grid, 256, 0, stream>>>(A16, W16, out, stats + 1);

  finalize_kernel<<<8192, 256, 0, stream>>>((float4*)out, nc4, stats);
}
